// SelfAttention_87582973100332
// MI455X (gfx1250) — matrix operand order repaired and checked by reading, not compiled, not run on hardware
//
#include <hip/hip_runtime.h>
#include <hip/hip_bf16.h>

#define DM    1024
#define NH    16
#define DH    64
#define BATCH 4
#define SEQ   2048
#define MROWS (BATCH*SEQ)   // 8192

typedef __attribute__((ext_vector_type(16))) __bf16 bf16x16;
typedef __attribute__((ext_vector_type(8)))  float  f32x8;

union FragAB {
    bf16x16 v;
    uint4   u[2];
    __bf16  h[16];
};

#ifndef __has_builtin
#define __has_builtin(x) 0
#endif
#if __has_builtin(__builtin_amdgcn_tensor_load_to_lds) && \
    __has_builtin(__builtin_amdgcn_s_wait_tensorcnt)
#define USE_TDM 1
#else
#define USE_TDM 0
#endif

#if USE_TDM
typedef __attribute__((ext_vector_type(4))) unsigned int u32x4;
typedef __attribute__((ext_vector_type(8))) int          i32x8;
typedef __attribute__((ext_vector_type(4))) int          i32x4;

// Issue one TDM descriptor moving a 128(rows) x 32(k) bf16 tile from global
// (row-major, row_stride elements) into contiguous LDS. ISA D# layout §8.3/8.4.
__device__ __forceinline__ void tdm_load_tile_128x32(
        const __bf16* gsrc, const __bf16* ldsdst,
        int tensor_rows, int row_stride) {
    unsigned long long ga = (unsigned long long)gsrc;
    unsigned int la = (unsigned int)(unsigned long long)ldsdst; // LDS aperture low bits = offset

    u32x4 g0;
    g0[0] = 1u;                                       // count=1, user descriptor
    g0[1] = la;                                       // lds_addr (bytes)
    g0[2] = (unsigned int)ga;                         // global_addr[31:0]
    g0[3] = ((unsigned int)(ga >> 32) & 0x01FFFFFFu)  // global_addr[56:32]
            | (2u << 30);                             // type = 2 ("image")

    i32x8 g1 = {};
    g1[0] = 1 << 16;                                  // data_size = 2 bytes
    g1[1] = (row_stride & 0xFFFF) << 16;              // tensor_dim0[15:0]  (bits 63:48)
    g1[2] = ((row_stride >> 16) & 0xFFFF)             // tensor_dim0[31:16] (bits 79:64)
            | ((tensor_rows & 0xFFFF) << 16);         // tensor_dim1[15:0]  (bits 111:80)
    g1[3] = ((tensor_rows >> 16) & 0xFFFF)            // tensor_dim1[31:16]
            | (32 << 16);                             // tile_dim0 = 32     (bits 127:112)
    g1[4] = 128;                                      // tile_dim1 = 128    (bits 143:128)
    g1[5] = row_stride;                               // tensor_dim0_stride[31:0]
    g1[6] = 0;
    g1[7] = 0;

    i32x4 gz = {};
#if __clang_major__ >= 23
    i32x8 gz8 = {};
    __builtin_amdgcn_tensor_load_to_lds(g0, g1, gz, gz, gz8, 0);
#else
    __builtin_amdgcn_tensor_load_to_lds(g0, g1, gz, gz, 0);
#endif
}
#endif // USE_TDM

// ---------------------------------------------------------------- f32 -> bf16
__global__ void cvt_f32_bf16(const float* __restrict__ src,
                             __bf16* __restrict__ dst, int n4) {
    int i = blockIdx.x * blockDim.x + threadIdx.x;
    if (i >= n4) return;
    float4 f = ((const float4*)src)[i];
    union { __bf16 h[4]; uint2 u; } o;
    o.h[0] = (__bf16)f.x; o.h[1] = (__bf16)f.y;
    o.h[2] = (__bf16)f.z; o.h[3] = (__bf16)f.w;
    ((uint2*)dst)[i] = o.u;
}

// ------------------------------------------------- Y = X @ W^T + b  (WMMA bf16)
// X:[M,K] bf16 row-major, W:[N,K] bf16 row-major (torch Linear weight layout).
// Block = 4 waves, block tile 128x128, wave tile 64x64 (4x4 WMMA 16x16x32).
// K-slices are streamed into LDS by the Tensor Data Mover, double-buffered.
template <bool OUT_BF16>
__global__ void __launch_bounds__(128)
gemm_bias(const __bf16* __restrict__ X, const __bf16* __restrict__ W,
          const float* __restrict__ bias, void* __restrict__ Yout,
          int M, int N, int K) {
    const int lane = threadIdx.x & 31;
    const int wave = threadIdx.x >> 5;
    const int wi = wave >> 1, wj = wave & 1;
    const int bm0 = blockIdx.x * 128;
    const int bn0 = blockIdx.y * 128;
    const int m0 = bm0 + wi * 64;
    const int n0 = bn0 + wj * 64;
    const int l15 = lane & 15;
    const int hi  = lane >> 4;

    const f32x8 zero = {};
    f32x8 acc[4][4];
#pragma unroll
    for (int i = 0; i < 4; ++i)
#pragma unroll
        for (int j = 0; j < 4; ++j) acc[i][j] = zero;

#if USE_TDM
    __shared__ __bf16 lA[2][128][32];
    __shared__ __bf16 lB[2][128][32];

    if (wave == 0) {
        tdm_load_tile_128x32(X + (size_t)bm0 * K, &lA[0][0][0], M, K);
        tdm_load_tile_128x32(W + (size_t)bn0 * K, &lB[0][0][0], N, K);
    }
    int buf = 0;
    for (int k0 = 0; k0 < K; k0 += 32, buf ^= 1) {
        const bool more = (k0 + 32) < K;
        if (more && wave == 0) {
            tdm_load_tile_128x32(X + (size_t)bm0 * K + (k0 + 32),
                                 &lA[buf ^ 1][0][0], M, K);
            tdm_load_tile_128x32(W + (size_t)bn0 * K + (k0 + 32),
                                 &lB[buf ^ 1][0][0], N, K);
        }
        // TDM ops from one wave complete in order: waiting down to 2 releases
        // exactly the current buffer's pair while the next pair is in flight.
        if (more) __builtin_amdgcn_s_wait_tensorcnt(2);
        else      __builtin_amdgcn_s_wait_tensorcnt(0);
        __syncthreads();

        FragAB af[4], bf[4];
#pragma unroll
        for (int i = 0; i < 4; ++i) {
            const __bf16* p = &lA[buf][wi * 64 + i * 16 + l15][hi * 16];
            af[i].u[0] = *(const uint4*)p;          // K = kb .. kb+7
            af[i].u[1] = *(const uint4*)(p + 8);    // K = kb+8 .. kb+15
        }
#pragma unroll
        for (int j = 0; j < 4; ++j) {
            const __bf16* p = &lB[buf][wj * 64 + j * 16 + l15][hi * 16];
            bf[j].u[0] = *(const uint4*)p;          // K = kb .. kb+7
            bf[j].u[1] = *(const uint4*)(p + 8);    // K = kb+8 .. kb+15
        }
#pragma unroll
        for (int i = 0; i < 4; ++i)
#pragma unroll
            for (int j = 0; j < 4; ++j)
                acc[i][j] = __builtin_amdgcn_wmma_f32_16x16x32_bf16(
                    false, af[i].v, false, bf[j].v, (short)0, acc[i][j],
                    false, false);
        __syncthreads();
    }
#else  // fallback: direct global fragment loads (whole working set is L2-resident)
    for (int k0 = 0; k0 < K; k0 += 32) {
        FragAB af[4], bf[4];
#pragma unroll
        for (int i = 0; i < 4; ++i) {
            const __bf16* p = X + (size_t)(m0 + i * 16 + l15) * K + k0 + hi * 16;
            af[i].u[0] = *(const uint4*)p;
            af[i].u[1] = *(const uint4*)(p + 8);
        }
#pragma unroll
        for (int j = 0; j < 4; ++j) {
            const __bf16* p = W + (size_t)(n0 + j * 16 + l15) * K + k0 + hi * 16;
            bf[j].u[0] = *(const uint4*)p;
            bf[j].u[1] = *(const uint4*)(p + 8);
        }
#pragma unroll
        for (int i = 0; i < 4; ++i)
#pragma unroll
            for (int j = 0; j < 4; ++j)
                acc[i][j] = __builtin_amdgcn_wmma_f32_16x16x32_bf16(
                    false, af[i].v, false, bf[j].v, (short)0, acc[i][j],
                    false, false);
    }
#endif

#pragma unroll
    for (int j = 0; j < 4; ++j) {
        const int n = n0 + j * 16 + l15;
        const float bj = bias[n];
#pragma unroll
        for (int i = 0; i < 4; ++i) {
#pragma unroll
            for (int r = 0; r < 8; ++r) {
                const int m = m0 + i * 16 + (hi ? 8 + r : r);
                const float v = acc[i][j][r] + bj;
                if (OUT_BF16)
                    ((__bf16*)Yout)[(size_t)m * N + n] = (__bf16)v;
                else
                    ((float*)Yout)[(size_t)m * N + n] = v;
            }
        }
    }
}

// -------------------------------------------- causal flash attention (bf16 WMMA)
// One wave handles 16 query rows of one (b,h); block = 4 waves = 64 rows.
__global__ void __launch_bounds__(128)
flash_attn(const __bf16* __restrict__ Qm, const __bf16* __restrict__ Km,
           const __bf16* __restrict__ Vm, __bf16* __restrict__ Om) {
    __shared__ __bf16 plds[4][16][32];   // per-wave P tile staging

    const int lane = threadIdx.x & 31;
    const int wave = threadIdx.x >> 5;
    const int l15  = lane & 15;
    const int hi   = lane >> 4;

    const int blk = blockIdx.x;          // b*NH*(SEQ/64) + h*(SEQ/64) + qb
    const int qb = blk & 31;             // SEQ/64 = 32
    const int h  = (blk >> 5) & (NH - 1);
    const int b  = blk >> 9;
    const int q0 = qb * 64 + wave * 16;

    const size_t rowbase = (size_t)b * SEQ;
    const int col0 = h * DH;

    // Q fragments for d-chunks {0,32}, loaded once and reused across all t.
    FragAB aq[2];
#pragma unroll
    for (int s = 0; s < 2; ++s) {
        const __bf16* p =
            Qm + (rowbase + q0 + l15) * DM + col0 + s * 32 + hi * 16;
        aq[s].u[0] = *(const uint4*)p;
        aq[s].u[1] = *(const uint4*)(p + 8);
    }

    const f32x8 zero = {};
    f32x8 oacc[4];
#pragma unroll
    for (int jd = 0; jd < 4; ++jd) oacc[jd] = zero;
    float mrow[8], lrow[8];
#pragma unroll
    for (int r = 0; r < 8; ++r) { mrow[r] = -3.0e38f; lrow[r] = 0.0f; }

    for (int t0 = 0; t0 <= q0 + 15; t0 += 32) {
        // ---- scores S = Q K^T (two 16x16 tiles over t, K-dim = 64 over d)
        f32x8 s[2]; s[0] = zero; s[1] = zero;
#pragma unroll
        for (int ds = 0; ds < 2; ++ds) {
            FragAB bk[2];
#pragma unroll
            for (int j = 0; j < 2; ++j) {
                const __bf16* p = Km + (rowbase + t0 + j * 16 + l15) * DM +
                                  col0 + ds * 32 + hi * 16;
                bk[j].u[0] = *(const uint4*)p;
                bk[j].u[1] = *(const uint4*)(p + 8);
            }
#pragma unroll
            for (int j = 0; j < 2; ++j)
                s[j] = __builtin_amdgcn_wmma_f32_16x16x32_bf16(
                    false, aq[ds].v, false, bk[j].v, (short)0, s[j],
                    false, false);
        }

        // ---- causal mask + online softmax (rows live in 16-lane halves)
        float pv[2][8];
#pragma unroll
        for (int r = 0; r < 8; ++r) {
            const int q = q0 + (hi ? 8 + r : r);
            float mx = -3.0e38f;
#pragma unroll
            for (int j = 0; j < 2; ++j) {
                const int t = t0 + j * 16 + l15;
                const float val = (t <= q) ? s[j][r] * 0.125f : -3.0e38f;
                pv[j][r] = val;
                mx = fmaxf(mx, val);
            }
#pragma unroll
            for (int d = 1; d < 16; d <<= 1)
                mx = fmaxf(mx, __shfl_xor(mx, d, 32));
            const float mnew  = fmaxf(mrow[r], mx);
            const float alpha = __expf(mrow[r] - mnew);
            mrow[r] = mnew;
            float psum = 0.0f;
#pragma unroll
            for (int j = 0; j < 2; ++j) {
                const float p = __expf(pv[j][r] - mnew);
                pv[j][r] = p;
                psum += p;
            }
#pragma unroll
            for (int d = 1; d < 16; d <<= 1) psum += __shfl_xor(psum, d, 32);
            lrow[r] = lrow[r] * alpha + psum;
#pragma unroll
            for (int jd = 0; jd < 4; ++jd) oacc[jd][r] *= alpha;
        }

        // ---- C-layout P -> LDS -> A-layout fragment (cross-lane reshuffle)
#pragma unroll
        for (int j = 0; j < 2; ++j)
#pragma unroll
            for (int r = 0; r < 8; ++r) {
                const int m = hi ? 8 + r : r;
                plds[wave][m][j * 16 + l15] = (__bf16)pv[j][r];
            }
        asm volatile("s_wait_dscnt 0" ::: "memory");
        FragAB ap;
        {
            const __bf16* p = &plds[wave][l15][hi * 16];
            ap.u[0] = *(const uint4*)p;
            ap.u[1] = *(const uint4*)(p + 8);
        }

        // ---- O += P @ V  (K-dim = 32 over t, 4 n-tiles over d)
#pragma unroll
        for (int jd = 0; jd < 4; ++jd) {
            FragAB bv;
#pragma unroll
            for (int e = 0; e < 16; ++e) {
                const int t = t0 + hi * 16 + e;
                bv.h[e] = Vm[(rowbase + t) * DM + col0 + jd * 16 + l15];
            }
            oacc[jd] = __builtin_amdgcn_wmma_f32_16x16x32_bf16(
                false, ap.v, false, bv.v, (short)0, oacc[jd], false, false);
        }
    }

    // ---- normalize and write O (bf16, [b*T+q, h*64+d] layout for next GEMM)
#pragma unroll
    for (int r = 0; r < 8; ++r) {
        const float inv = 1.0f / lrow[r];
        const int q = q0 + (hi ? 8 + r : r);
#pragma unroll
        for (int jd = 0; jd < 4; ++jd)
            Om[(rowbase + q) * DM + col0 + jd * 16 + l15] =
                (__bf16)(oacc[jd][r] * inv);
    }
}

// ------------------------------------------------------------------- launcher
extern "C" void kernel_launch(void* const* d_in, const int* in_sizes, int n_in,
                              void* d_out, int out_size, void* d_ws,
                              size_t ws_size, hipStream_t stream) {
    const float* x  = (const float*)d_in[0];
    const float* Wq = (const float*)d_in[1];
    const float* bq = (const float*)d_in[2];
    const float* Wk = (const float*)d_in[3];
    const float* bk = (const float*)d_in[4];
    const float* Wv = (const float*)d_in[5];
    const float* bv = (const float*)d_in[6];
    const float* Wo = (const float*)d_in[7];
    const float* bo = (const float*)d_in[8];

    char* ws = (char*)d_ws;
    __bf16* xb  = (__bf16*)ws; ws += (size_t)MROWS * DM * 2;
    __bf16* wqb = (__bf16*)ws; ws += (size_t)DM * DM * 2;
    __bf16* wkb = (__bf16*)ws; ws += (size_t)DM * DM * 2;
    __bf16* wvb = (__bf16*)ws; ws += (size_t)DM * DM * 2;
    __bf16* wob = (__bf16*)ws; ws += (size_t)DM * DM * 2;
    __bf16* Qb  = (__bf16*)ws; ws += (size_t)MROWS * DM * 2;
    __bf16* Kb  = (__bf16*)ws; ws += (size_t)MROWS * DM * 2;
    __bf16* Vb  = (__bf16*)ws; ws += (size_t)MROWS * DM * 2;
    __bf16* Ob  = (__bf16*)ws; ws += (size_t)MROWS * DM * 2;

    auto cvt = [&](const float* s, __bf16* d, size_t n) {
        int n4 = (int)(n / 4);
        cvt_f32_bf16<<<(n4 + 255) / 256, 256, 0, stream>>>(s, d, n4);
    };
    cvt(x,  xb,  (size_t)MROWS * DM);
    cvt(Wq, wqb, (size_t)DM * DM);
    cvt(Wk, wkb, (size_t)DM * DM);
    cvt(Wv, wvb, (size_t)DM * DM);
    cvt(Wo, wob, (size_t)DM * DM);

    dim3 gg(MROWS / 128, DM / 128);
    gemm_bias<true><<<gg, 128, 0, stream>>>(xb, wqb, bq, Qb, MROWS, DM, DM);
    gemm_bias<true><<<gg, 128, 0, stream>>>(xb, wkb, bk, Kb, MROWS, DM, DM);
    gemm_bias<true><<<gg, 128, 0, stream>>>(xb, wvb, bv, Vb, MROWS, DM, DM);

    flash_attn<<<BATCH * NH * (SEQ / 64), 128, 0, stream>>>(Qb, Kb, Vb, Ob);

    gemm_bias<false><<<gg, 128, 0, stream>>>(Ob, wob, bo, d_out, MROWS, DM, DM);
}
